// TopkActMlp_47304769798196
// MI455X (gfx1250) — matrix operand order repaired and checked by reading, not compiled, not run on hardware
//
#include <hip/hip_runtime.h>
#include <math.h>

// ---------------------------------------------------------------------------
// TopK-masked MLP for MI455X (gfx1250, wave32, WMMA + async-LDS pipeline)
//   h = x @ W1 + b1 ; h = gelu(topk_mask(h, 409)) ; y = h @ W2 + b2
//   out = topk_mask(y, 102)
// GEMMs: v_wmma_f32_16x16x32_bf16, operands double-buffered into LDS with
// global_load_async_to_lds_b128 (ASYNCcnt), shared by all 4 waves of a block.
// ---------------------------------------------------------------------------

typedef __attribute__((ext_vector_type(16))) __bf16         v16bf;
typedef __attribute__((ext_vector_type(8)))  float          v8f;
typedef __attribute__((ext_vector_type(8)))  unsigned short u16x8;
typedef __attribute__((ext_vector_type(4)))  float          f32x4;

#define D_MODEL 1024
#define D_HID   4096
#define ROWS    16384   // 8 * 2048
#define K1      409     // topk on hidden (4096 * 0.1)
#define K2      102     // topk on output (1024 * 0.1)

__device__ __forceinline__ unsigned short f2bf(float f) {
  unsigned u = __float_as_uint(f);
  u += 0x7FFFu + ((u >> 16) & 1u);   // round-to-nearest-even
  return (unsigned short)(u >> 16);
}

// order-preserving float -> uint key (larger key == larger float)
__device__ __forceinline__ unsigned fkey(float f) {
  unsigned u = __float_as_uint(f);
  return (u & 0x80000000u) ? ~u : (u | 0x80000000u);
}

// ---------------------------------------------------------------------------
// Repack f32 row-major [M x K] into WMMA A-fragment tiles (16x32 bf16, 1KB).
// Tile (mt,kt), lane l holds M = mt*16 + (l&15);
//   lanes 0-15 : K = {0..7, 16..23},  lanes 16-31 : K = {8..15, 24..31}
// stored lane-major: lane's 16 bf16 contiguous at tile*512 + lane*16.
// ---------------------------------------------------------------------------
__global__ void pack_a_f32(const float* __restrict__ src, unsigned short* __restrict__ dst,
                           int K, int Ktiles) {
  int tile = blockIdx.x * 8 + (threadIdx.x >> 5);
  int lane = threadIdx.x & 31;
  int mt = tile / Ktiles, kt = tile % Ktiles;
  int m = mt * 16 + (lane & 15);
  int kbase = kt * 32 + ((lane >> 4) << 3);     // +8 for hi half-wave
  const float* row = src + (size_t)m * K;
  unsigned short v[16];
#pragma unroll
  for (int s = 0; s < 16; ++s) {
    int k = kbase + ((s < 8) ? s : (s + 8));
    v[s] = f2bf(row[k]);
  }
  u16x8* out = (u16x8*)(dst + (size_t)tile * 512 + lane * 16);
  out[0] = *(u16x8*)&v[0];
  out[1] = *(u16x8*)&v[8];
}

// ---------------------------------------------------------------------------
// Repack f32 row-major [K x N] into WMMA B-fragment tiles (32x16 bf16, 1KB).
// Tile (nt,kt), lane l holds N = nt*16 + (l&15);
//   lanes 0-15 : K = kt*32 + {0..7, 16..23}, lanes 16-31 : K = kt*32 + {8..15, 24..31}
// ---------------------------------------------------------------------------
__global__ void pack_b_f32(const float* __restrict__ src, unsigned short* __restrict__ dst,
                           int N, int Ktiles) {
  int tile = blockIdx.x * 8 + (threadIdx.x >> 5);   // tile = nt*Ktiles + kt
  int lane = threadIdx.x & 31;
  int nt = tile / Ktiles, kt = tile % Ktiles;
  int n = nt * 16 + (lane & 15);
  int kbase = kt * 32 + ((lane >> 4) << 3);         // +8 for hi half-wave
  unsigned short v[16];
#pragma unroll
  for (int s = 0; s < 16; ++s)
    v[s] = f2bf(src[(size_t)(kbase + ((s < 8) ? s : (s + 8))) * N + n]);
  u16x8* out = (u16x8*)(dst + (size_t)tile * 512 + lane * 16);
  out[0] = *(u16x8*)&v[0];
  out[1] = *(u16x8*)&v[8];
}

// ---------------------------------------------------------------------------
// BF16 WMMA GEMM: C[M x N] = A[M x K] * B[K x N] + bias (f32 accumulate).
// Block = 128 threads = 4 waves (2x2); wave computes a 64x64 C tile
// (4x4 grid of 16x16 WMMA tiles, 128 accumulator VGPRs).
// Per K-chunk the block stages 8 A-tiles + 8 B-tiles (16KB) into LDS with
// async loads (double buffered), so each fragment is fetched from L2 once
// per block instead of twice.
// ---------------------------------------------------------------------------
__global__ __launch_bounds__(128) void gemm_bf16_wmma(
    const unsigned short* __restrict__ Ap, const unsigned short* __restrict__ Bp,
    const float* __restrict__ bias, float* __restrict__ C, int Ktiles, int N) {
  __shared__ unsigned short smem[2][8192];   // [buf][16 tiles * 512 bf16] = 2 x 16KB

  int lane = threadIdx.x & 31;
  int wave = threadIdx.x >> 5;
  int mtb = blockIdx.x * 8;
  int ntb = blockIdx.y * 8;

  // Staging: wave w stages slots 4w..4w+3. Slot 0-7 = A tile (mtb+slot),
  // slot 8-15 = B tile (ntb+slot-8). Each lane async-copies its 32-byte
  // fragment run (global and LDS layouts match: tile*1KB + lane*32B).
  const size_t rowstride = (size_t)Ktiles * 512;
  const unsigned short* srcbase[4];
#pragma unroll
  for (int s = 0; s < 4; ++s) {
    int slot = wave * 4 + s;
    srcbase[s] = (slot < 8) ? (Ap + (size_t)(mtb + slot) * rowstride + lane * 16)
                            : (Bp + (size_t)(ntb + (slot - 8)) * rowstride + lane * 16);
  }

  auto stage = [&](int kt, int buf) {
#pragma unroll
    for (int s = 0; s < 4; ++s) {
      int slot = wave * 4 + s;
      unsigned long long g = (unsigned long long)(uintptr_t)(srcbase[s] + (size_t)kt * 512);
      unsigned l = (unsigned)(uintptr_t)(&smem[buf][slot * 512 + lane * 16]);
      asm volatile("global_load_async_to_lds_b128 %0, %1, off\n\t"
                   "global_load_async_to_lds_b128 %0, %1, off offset:16"
                   :: "v"(l), "v"(g) : "memory");
    }
  };

  stage(0, 0);

  v8f acc[4][4] = {};
  int aslot0 = (wave >> 1) * 4;        // this wave's 4 A slots
  int bslot0 = 8 + (wave & 1) * 4;     // this wave's 4 B slots

  for (int kt = 0; kt < Ktiles; ++kt) {
    int buf = kt & 1;
    asm volatile("s_wait_asynccnt 0x0" ::: "memory");
    __syncthreads();                   // buffer `buf` fully staged by all waves

    if (kt + 1 < Ktiles) stage(kt + 1, buf ^ 1);

    // prefetch two chunks ahead (global_prefetch_b8)
    __builtin_prefetch((const void*)(srcbase[0] + (size_t)(kt + 2) * 512), 0, 1);
    __builtin_prefetch((const void*)(srcbase[2] + (size_t)(kt + 2) * 512), 0, 1);

    v16bf a[4], b[4];
#pragma unroll
    for (int i = 0; i < 4; ++i)
      a[i] = *(const v16bf*)(&smem[buf][(aslot0 + i) * 512 + lane * 16]);
#pragma unroll
    for (int j = 0; j < 4; ++j)
      b[j] = *(const v16bf*)(&smem[buf][(bslot0 + j) * 512 + lane * 16]);

#pragma unroll
    for (int i = 0; i < 4; ++i)
#pragma unroll
      for (int j = 0; j < 4; ++j)
        acc[i][j] = __builtin_amdgcn_wmma_f32_16x16x32_bf16(
            false, a[i], false, b[j], (short)0, acc[i][j], false, false);

    __syncthreads();                   // all waves done reading `buf`
  }

  // Epilogue: C/D layout — lanes 0-15: M=r, N=lane ; lanes 16-31: M=r+8
  int mt0 = mtb + (wave >> 1) * 4;
  int nt0 = ntb + (wave & 1) * 4;
  int ncol  = lane & 15;
  int mrow0 = (lane < 16) ? 0 : 8;
#pragma unroll
  for (int j = 0; j < 4; ++j) {
    int col = (nt0 + j) * 16 + ncol;
    float bv = bias[col];
#pragma unroll
    for (int i = 0; i < 4; ++i) {
      int row0 = (mt0 + i) * 16 + mrow0;
#pragma unroll
      for (int r = 0; r < 8; ++r)
        C[(size_t)(row0 + r) * N + col] = acc[i][j][r] + bv;
    }
  }
}

// ---------------------------------------------------------------------------
// Per-row (4096) exact top-k threshold via 4-round radix select (8-bit LDS
// histogram on order-preserving keys), then mask + exact GELU, writing bf16
// directly in WMMA A-fragment layout for GEMM2 (two contiguous 16B stores
// per thread — no separate repack pass).
// ---------------------------------------------------------------------------
__global__ __launch_bounds__(256) void topk_gelu_pack(
    const float* __restrict__ H, unsigned short* __restrict__ Ap2, int KtilesOut) {
  __shared__ unsigned hist[256];
  __shared__ unsigned sPfx, sKr;
  int m = blockIdx.x;
  int t = threadIdx.x;
  const float* row = H + (size_t)m * D_HID;

  float v[16];
  unsigned key[16];
#pragma unroll
  for (int s = 0; s < 16; ++s) {
    v[s] = row[t * 16 + s];
    key[s] = fkey(v[s]);
  }

  unsigned prefix = 0, kr = K1;
  for (int shift = 24; shift >= 0; shift -= 8) {
    hist[t] = 0;
    __syncthreads();
    unsigned himask = (shift == 24) ? 0u : (0xFFFFFFFFu << (shift + 8));
#pragma unroll
    for (int s = 0; s < 16; ++s)
      if ((key[s] & himask) == prefix)
        atomicAdd(&hist[(key[s] >> shift) & 255u], 1u);
    __syncthreads();
    if (t == 0) {
      unsigned cum = 0;
      int b = 255;
      for (; b > 0; --b) {
        unsigned c = hist[b];
        if (cum + c >= kr) break;
        cum += c;
      }
      sPfx = prefix | ((unsigned)b << shift);
      sKr = kr - cum;
    }
    __syncthreads();
    prefix = sPfx;
    kr = sKr;
    __syncthreads();
  }
  unsigned T = prefix;  // exact key of the k-th largest element

  unsigned short o[16];
#pragma unroll
  for (int s = 0; s < 16; ++s) {
    float g = 0.0f;
    if (key[s] >= T) {
      float x = v[s];
      g = 0.5f * x * (1.0f + erff(x * 0.70710678118654752440f));  // exact GELU
    }
    o[s] = f2bf(g);
  }

  // scatter into A-fragment layout: thread covers k in [t*16, t*16+16)
  int mt = m >> 4, mm = m & 15;
  int kt = t >> 1, half = t & 1;
  size_t tbase = ((size_t)mt * KtilesOut + kt) * 512;
  *(u16x8*)(Ap2 + tbase + (size_t)mm * 16 + half * 8)        = *(u16x8*)&o[0];
  *(u16x8*)(Ap2 + tbase + (size_t)(mm + 16) * 16 + half * 8) = *(u16x8*)&o[8];
}

// ---------------------------------------------------------------------------
// Final per-row (1024) top-k (k=102) mask, f32 output.
// ---------------------------------------------------------------------------
__global__ __launch_bounds__(256) void topk_out(const float* __restrict__ Y,
                                                float* __restrict__ O) {
  __shared__ unsigned hist[256];
  __shared__ unsigned sPfx, sKr;
  int m = blockIdx.x;
  int t = threadIdx.x;
  const float* row = Y + (size_t)m * D_MODEL;

  float v[4];
  unsigned key[4];
#pragma unroll
  for (int s = 0; s < 4; ++s) {
    v[s] = row[t * 4 + s];
    key[s] = fkey(v[s]);
  }

  unsigned prefix = 0, kr = K2;
  for (int shift = 24; shift >= 0; shift -= 8) {
    hist[t] = 0;
    __syncthreads();
    unsigned himask = (shift == 24) ? 0u : (0xFFFFFFFFu << (shift + 8));
#pragma unroll
    for (int s = 0; s < 4; ++s)
      if ((key[s] & himask) == prefix)
        atomicAdd(&hist[(key[s] >> shift) & 255u], 1u);
    __syncthreads();
    if (t == 0) {
      unsigned cum = 0;
      int b = 255;
      for (; b > 0; --b) {
        unsigned c = hist[b];
        if (cum + c >= kr) break;
        cum += c;
      }
      sPfx = prefix | ((unsigned)b << shift);
      sKr = kr - cum;
    }
    __syncthreads();
    prefix = sPfx;
    kr = sKr;
    __syncthreads();
  }
  unsigned T = prefix;

  f32x4 o;
#pragma unroll
  for (int s = 0; s < 4; ++s) o[s] = (key[s] >= T) ? v[s] : 0.0f;
  *(f32x4*)(O + (size_t)m * D_MODEL + t * 4) = o;
}

// ---------------------------------------------------------------------------
extern "C" void kernel_launch(void* const* d_in, const int* in_sizes, int n_in,
                              void* d_out, int out_size, void* d_ws, size_t ws_size,
                              hipStream_t stream) {
  const float* x  = (const float*)d_in[0];
  const float* W1 = (const float*)d_in[1];
  const float* b1 = (const float*)d_in[2];
  const float* W2 = (const float*)d_in[3];
  const float* b2 = (const float*)d_in[4];
  float* out = (float*)d_out;

  char* ws = (char*)d_ws;
  size_t off = 0;
  auto alloc = [&](size_t bytes) -> void* {
    void* p = ws + off;
    off += (bytes + 1023) & ~(size_t)1023;
    return p;
  };
  unsigned short* Ap1 = (unsigned short*)alloc((size_t)ROWS * D_MODEL * 2);  // x bf16 A-tiles
  unsigned short* Bp1 = (unsigned short*)alloc((size_t)D_MODEL * D_HID * 2); // W1 bf16 B-tiles
  unsigned short* Bp2 = (unsigned short*)alloc((size_t)D_HID * D_MODEL * 2); // W2 bf16 B-tiles
  float*          Hb  = (float*)alloc((size_t)ROWS * D_HID * 4);             // fc1 out f32
  unsigned short* Ap2 = (unsigned short*)alloc((size_t)ROWS * D_HID * 2);    // masked gelu bf16 A-tiles
  float*          Yb  = (float*)alloc((size_t)ROWS * D_MODEL * 4);           // fc2 out f32

  // repack inputs/weights into WMMA fragment layouts (bf16)
  pack_a_f32<<<(ROWS / 16) * (D_MODEL / 32) / 8, 256, 0, stream>>>(x, Ap1, D_MODEL, D_MODEL / 32);
  pack_b_f32<<<(D_HID / 16) * (D_MODEL / 32) / 8, 256, 0, stream>>>(W1, Bp1, D_HID, D_MODEL / 32);
  pack_b_f32<<<(D_MODEL / 16) * (D_HID / 32) / 8, 256, 0, stream>>>(W2, Bp2, D_MODEL, D_HID / 32);

  // fc1: [16384 x 1024] * [1024 x 4096] + b1
  gemm_bf16_wmma<<<dim3(ROWS / 128, D_HID / 128), 128, 0, stream>>>(
      Ap1, Bp1, b1, Hb, D_MODEL / 32, D_HID);

  // topk(409) + exact gelu + pack to A-fragments for fc2
  topk_gelu_pack<<<ROWS, 256, 0, stream>>>(Hb, Ap2, D_HID / 32);

  // fc2: [16384 x 4096] * [4096 x 1024] + b2
  gemm_bf16_wmma<<<dim3(ROWS / 128, D_MODEL / 128), 128, 0, stream>>>(
      Ap2, Bp2, b2, Yb, D_HID / 32, D_MODEL);

  // final topk(102) mask
  topk_out<<<ROWS, 256, 0, stream>>>(Yb, out);
}
